// SparseAttention_23167053595226
// MI455X (gfx1250) — matrix operand order repaired and checked by reading, not compiled, not run on hardware
//
#include <hip/hip_runtime.h>
#include <hip/hip_bf16.h>
#include <stdint.h>

typedef __attribute__((ext_vector_type(16))) _Float16 v16h;
typedef __attribute__((ext_vector_type(8)))  float    v8f;
typedef __attribute__((__vector_size__(16))) int      v4i;

#define B_   4
#define S_   2048
#define D_   1024
#define H_   16
#define DH_  64
#define HD_  (H_ * DH_)   // 1024
#define BLK_ 64           // sparse block == SPARSE_FACTOR

#define GLOBAL_AS __attribute__((address_space(1)))
#define LDS_AS    __attribute__((address_space(3)))

#if __has_builtin(__builtin_amdgcn_global_load_async_to_lds_b128)
#define ASYNC_LDS 1
#else
#define ASYNC_LDS 0
#endif

// 16-byte global -> LDS copy; async (ASYNCcnt) when the gfx1250 builtin exists.
__device__ __forceinline__ void copy16_g2l(_Float16* dst_lds, const _Float16* src) {
#if ASYNC_LDS
  GLOBAL_AS v4i* g = (GLOBAL_AS v4i*)(GLOBAL_AS void*)(void*)const_cast<_Float16*>(src);
  LDS_AS    v4i* l = (LDS_AS    v4i*)(LDS_AS    void*)(void*)dst_lds;
  __builtin_amdgcn_global_load_async_to_lds_b128(g, l, 0, 0);
#else
  *reinterpret_cast<uint4*>(dst_lds) = *reinterpret_cast<const uint4*>(src);
#endif
}

__device__ __forceinline__ void wait_async_lds() {
#if ASYNC_LDS
  asm volatile("s_wait_asynccnt 0x0" ::: "memory");
#endif
}

// ---------------------------------------------------------------------------
// WMMA fragment loaders from LDS, per CDNA5 ISA 7.12.2 (16-bit layouts, wave32)
// A 16x32: lane m=lane%16; VGPR r holds K pair {0,2,4,6,16,18,20,22}[r] + 8*(lane/16)
// B 32x16: lane n=lane%16; element s holds K {0..7,16..23}[s] + 8*(lane/16)
// ---------------------------------------------------------------------------
template<int PITCH>
__device__ __forceinline__ v16h frag_a(const _Float16* base, int m0, int k0, int lane) {
  const int hs = (lane >> 4) & 1;
  const int lm = lane & 15;
  const _Float16* row = base + (m0 + lm) * PITCH + k0 + 8 * hs;
  v16h f;
#pragma unroll
  for (int r = 0; r < 8; ++r) {
    const int kk = (r < 4) ? (2 * r) : (16 + 2 * (r - 4));
    f[2 * r]     = row[kk];
    f[2 * r + 1] = row[kk + 1];
  }
  return f;
}

template<int PITCH>
__device__ __forceinline__ v16h frag_b(const _Float16* base, int n0, int k0, int lane) {
  const int hs = (lane >> 4) & 1;
  const int lm = lane & 15;
  const _Float16* row = base + (n0 + lm) * PITCH + k0 + 8 * hs;
  v16h f;
#pragma unroll
  for (int r = 0; r < 16; ++r) f[r] = row[(r < 8) ? r : (r + 8)];
  return f;
}

// ---------------------------------------------------------------------------
// One-time precision conversion kernels
// ---------------------------------------------------------------------------
__global__ __launch_bounds__(256)
void convert_f32_to_f16(const float* __restrict__ src, _Float16* __restrict__ dst, int n4) {
  const int i = blockIdx.x * blockDim.x + threadIdx.x;
  if (i < n4) {
    const float4 v = reinterpret_cast<const float4*>(src)[i];
    _Float16* d = dst + 4 * (size_t)i;
    d[0] = (_Float16)v.x; d[1] = (_Float16)v.y;
    d[2] = (_Float16)v.z; d[3] = (_Float16)v.w;
  }
}

// W[K,N] f32 -> Wt[N,K] f16 (32x32 LDS tile transpose, coalesced both sides)
__global__ __launch_bounds__(256)
void transpose_w_f16(const float* __restrict__ W, _Float16* __restrict__ Wt,
                     int K, int N) {
  __shared__ _Float16 t[32][33];
  const int n0 = blockIdx.x * 32, k0 = blockIdx.y * 32;
  const int tx = threadIdx.x & 31, ty = threadIdx.x >> 5;   // ty 0..7
#pragma unroll
  for (int i = 0; i < 4; ++i)
    t[ty + 8 * i][tx] = (_Float16)W[(size_t)(k0 + ty + 8 * i) * N + n0 + tx];
  __syncthreads();
#pragma unroll
  for (int i = 0; i < 4; ++i)
    Wt[(size_t)(n0 + ty + 8 * i) * K + k0 + tx] = t[tx][ty + 8 * i];
}

// ---------------------------------------------------------------------------
// GEMM: C[M,N] = A[M,K] (f16) @ Bt[N,K]^T (f16) + bias[N], f32 accumulate.
// 128x128 block tile, BK=64, double-buffered async LDS staging.
// 256 threads = 8 waves in 4(row) x 2(col) grid; 2x4 16x16 accs per wave.
// ---------------------------------------------------------------------------
template<typename OutT>
__global__ __launch_bounds__(256)
void gemm_wmma(const _Float16* __restrict__ A, const _Float16* __restrict__ Bt,
               const float* __restrict__ bias, OutT* __restrict__ C,
               int M, int N, int K) {
  constexpr int BM = 128, BN = 128, BK = 64, PITCH = BK + 8;  // 72 halfs = 144B
  __shared__ _Float16 sA[2][BM * PITCH];
  __shared__ _Float16 sB[2][BN * PITCH];

  const int tid  = threadIdx.x;
  const int lane = tid & 31;
  const int wave = tid >> 5;
  const int wm   = wave & 3;    // 4 x 32-row groups
  const int wn   = wave >> 2;   // 2 x 64-col groups
  const int rowBase = blockIdx.y * BM;
  const int colBase = blockIdx.x * BN;

  v8f acc[2][4];
  const v8f vzero = {};
#pragma unroll
  for (int i = 0; i < 2; ++i)
#pragma unroll
    for (int j = 0; j < 4; ++j) acc[i][j] = vzero;

  // stage one BK-slab of A and Bt into LDS buffer `buf` (16B chunks)
  auto stage = [&](int buf, int k0) {
#pragma unroll
    for (int i = 0; i < 4; ++i) {
      const int cc  = tid + 256 * i;      // 0..1023
      const int row = cc >> 3;            // 0..127
      const int seg = (cc & 7) * 8;       // half offset within 128B row
      copy16_g2l(&sA[buf][row * PITCH + seg],
                 &A[(size_t)(rowBase + row) * K + k0 + seg]);
      copy16_g2l(&sB[buf][row * PITCH + seg],
                 &Bt[(size_t)(colBase + row) * K + k0 + seg]);
    }
  };

  stage(0, 0);
  int buf = 0;
  for (int k0 = 0; k0 < K; k0 += BK, buf ^= 1) {
    wait_async_lds();       // our stage(buf) writes have landed
    __syncthreads();        // everyone's have; previous compute done
    if (k0 + BK < K) stage(buf ^ 1, k0 + BK);   // prefetch next slab

#pragma unroll
    for (int ks = 0; ks < BK; ks += 32) {
      v16h af[2], bf[4];
#pragma unroll
      for (int i = 0; i < 2; ++i) af[i] = frag_a<PITCH>(sA[buf], wm * 32 + i * 16, ks, lane);
#pragma unroll
      for (int j = 0; j < 4; ++j) bf[j] = frag_b<PITCH>(sB[buf], wn * 64 + j * 16, ks, lane);
#pragma unroll
      for (int i = 0; i < 2; ++i)
#pragma unroll
        for (int j = 0; j < 4; ++j)
          acc[i][j] = __builtin_amdgcn_wmma_f32_16x16x32_f16(
              false, af[i], false, bf[j], (short)0, acc[i][j], false, false);
    }
  }

  // epilogue: C/D layout element (m = r + 8*(lane/16), n = lane%16)
  const int hs = (lane >> 4) & 1;
  const int lm = lane & 15;
#pragma unroll
  for (int i = 0; i < 2; ++i)
#pragma unroll
    for (int j = 0; j < 4; ++j)
#pragma unroll
      for (int r = 0; r < 8; ++r) {
        const int gm = rowBase + wm * 32 + i * 16 + r + 8 * hs;
        const int gn = colBase + wn * 64 + j * 16 + lm;
        C[(size_t)gm * N + gn] = (OutT)(acc[i][j][r] + bias[gn]);
      }
}

// ---------------------------------------------------------------------------
// Block-diagonal attention: one workgroup per (b, h, 64-query block).
// scores = QK^T/8 (masked), softmax over 64 in-block keys, ctx = P @ V (f16).
// ---------------------------------------------------------------------------
__global__ __launch_bounds__(256)
void block_attn(const _Float16* __restrict__ Q, const _Float16* __restrict__ Kh,
                const _Float16* __restrict__ V, const unsigned char* __restrict__ mask,
                _Float16* __restrict__ ctx) {
  constexpr int P2 = DH_ + 8;    // 72-half pitch (144B, 16B multiple)
  constexpr int PS = BLK_ + 4;
  __shared__ _Float16 sQ[BLK_ * P2];    // [q][d]
  __shared__ _Float16 sK[BLK_ * P2];    // [kk][d]  (rows are B-frags for QK^T)
  __shared__ _Float16 sVt[DH_ * P2];    // [d][kk]  (rows are B-frags for P@V)
  __shared__ float    sS[BLK_ * PS];    // scores / exp
  __shared__ _Float16 sP[BLK_ * P2];    // probs [q][kk]

  const int tid  = threadIdx.x;
  const int lane = tid & 31;
  const int wave = tid >> 5;
  const int q0 = blockIdx.x * BLK_;
  const int h  = blockIdx.y;
  const int b  = blockIdx.z;

  // ---- stage Q,K rows via async 16B chunks; V transposed manually ----
  {
#pragma unroll
    for (int i = 0; i < 2; ++i) {
      const int cc  = tid + 256 * i;     // 0..511
      const int row = cc >> 3;           // 0..63
      const int seg = (cc & 7) * 8;      // half offset in 128B row
      const size_t gb = ((size_t)(b * S_ + q0 + row)) * HD_ + h * DH_ + seg;
      copy16_g2l(&sQ[row * P2 + seg], &Q[gb]);
      copy16_g2l(&sK[row * P2 + seg], &Kh[gb]);
    }
    const int row = tid >> 2;            // 0..63
    const int seg = (tid & 3) * 16;      // 0,16,32,48
    const size_t gb = ((size_t)(b * S_ + q0 + row)) * HD_ + h * DH_ + seg;
#pragma unroll
    for (int i = 0; i < 16; ++i) sVt[(seg + i) * P2 + row] = V[gb + i];
  }
  wait_async_lds();
  __syncthreads();

  const int hs = (lane >> 4) & 1;
  const int lm = lane & 15;

  // ---- scores = Q @ K^T * (1/sqrt(64)), masked ----
#pragma unroll
  for (int t = 0; t < 2; ++t) {
    const int tile = wave * 2 + t;
    const int tm = (tile >> 2) * 16;
    const int tn = (tile & 3) * 16;
    v8f acc = {};
#pragma unroll
    for (int ks = 0; ks < DH_; ks += 32) {
      v16h aq = frag_a<P2>(sQ, tm, ks, lane);
      v16h bk = frag_b<P2>(sK, tn, ks, lane);   // K rows == K^T columns
      acc = __builtin_amdgcn_wmma_f32_16x16x32_f16(
          false, aq, false, bk, (short)0, acc, false, false);
    }
#pragma unroll
    for (int r = 0; r < 8; ++r) {
      const int qi = tm + r + 8 * hs;
      const int ki = tn + lm;
      const size_t midx = ((size_t)b * S_ + (q0 + qi)) * S_ + (q0 + ki);
      const float sc = acc[r] * 0.125f;
      sS[qi * PS + ki] = mask[midx] ? sc : -1.0e9f;
    }
  }
  __syncthreads();

  // ---- softmax over 64 keys: one row per thread ----
  if (tid < BLK_) {
    float mx = -3.4e38f;
    for (int k = 0; k < BLK_; ++k) mx = fmaxf(mx, sS[tid * PS + k]);
    float sum = 0.0f;
    for (int k = 0; k < BLK_; ++k) {
      const float e = __expf(sS[tid * PS + k] - mx);
      sum += e;
      sS[tid * PS + k] = e;
    }
    const float inv = 1.0f / sum;
    for (int k = 0; k < BLK_; ++k)
      sP[tid * P2 + k] = (_Float16)(sS[tid * PS + k] * inv);
  }
  __syncthreads();

  // ---- ctx = P @ V ----
#pragma unroll
  for (int t = 0; t < 2; ++t) {
    const int tile = wave * 2 + t;
    const int tm = (tile >> 2) * 16;
    const int tn = (tile & 3) * 16;
    v8f acc = {};
#pragma unroll
    for (int ks = 0; ks < BLK_; ks += 32) {
      v16h ap = frag_a<P2>(sP, tm, ks, lane);
      v16h bv = frag_b<P2>(sVt, tn, ks, lane);
      acc = __builtin_amdgcn_wmma_f32_16x16x32_f16(
          false, ap, false, bv, (short)0, acc, false, false);
    }
#pragma unroll
    for (int r = 0; r < 8; ++r) {
      const int qi = tm + r + 8 * hs;
      const int dn = tn + lm;
      ctx[((size_t)(b * S_ + q0 + qi)) * HD_ + h * DH_ + dn] = (_Float16)acc[r];
    }
  }
}

// ---------------------------------------------------------------------------
extern "C" void kernel_launch(void* const* d_in, const int* in_sizes, int n_in,
                              void* d_out, int out_size, void* d_ws, size_t ws_size,
                              hipStream_t stream) {
  const float* xq = (const float*)d_in[0];
  const float* xk = (const float*)d_in[1];
  const float* xv = (const float*)d_in[2];
  const unsigned char* mask = (const unsigned char*)d_in[3];  // bool [B,S,S]
  const float* Wq = (const float*)d_in[4];
  const float* bq = (const float*)d_in[5];
  const float* Wk = (const float*)d_in[6];
  const float* bk = (const float*)d_in[7];
  const float* Wv = (const float*)d_in[8];
  const float* bv = (const float*)d_in[9];
  const float* Wo = (const float*)d_in[10];
  const float* bo = (const float*)d_in[11];

  // ws layout (f16 elements):
  //   xqh|xkh|xvh : 3*e      Wt q|k|v|o : 4*(D*HD)      Qh|Kh|Vh : 3*e
  //   ctx (f16, e) aliases xqh (dead after the Q projection). Total ~109 MB.
  const size_t e = (size_t)B_ * S_ * HD_;          // 8388608
  const size_t w = (size_t)D_ * HD_;               // 1048576
  _Float16* xqh = (_Float16*)d_ws;
  _Float16* xkh = xqh + e;
  _Float16* xvh = xkh + e;
  _Float16* Wqt = xvh + e;
  _Float16* Wkt = Wqt + w;
  _Float16* Wvt = Wkt + w;
  _Float16* Wot = Wvt + w;
  _Float16* Qh  = Wot + w;
  _Float16* Kf  = Qh + e;
  _Float16* Vf  = Kf + e;
  _Float16* ctx = xqh;                             // alias

  const int M = B_ * S_;                           // 8192
  dim3 blk(256);

  // one-time conversions
  const int n4 = (int)(e / 4);
  convert_f32_to_f16<<<dim3((n4 + 255) / 256), blk, 0, stream>>>(xq, xqh, n4);
  convert_f32_to_f16<<<dim3((n4 + 255) / 256), blk, 0, stream>>>(xk, xkh, n4);
  convert_f32_to_f16<<<dim3((n4 + 255) / 256), blk, 0, stream>>>(xv, xvh, n4);
  dim3 gt(HD_ / 32, D_ / 32);
  transpose_w_f16<<<gt, blk, 0, stream>>>(Wq, Wqt, D_, HD_);
  transpose_w_f16<<<gt, blk, 0, stream>>>(Wk, Wkt, D_, HD_);
  transpose_w_f16<<<gt, blk, 0, stream>>>(Wv, Wvt, D_, HD_);
  transpose_w_f16<<<gt, blk, 0, stream>>>(Wo, Wot, HD_, D_);

  // projections (f16 in, f16 out, f32 accum)
  dim3 gp(HD_ / 128, M / 128);
  gemm_wmma<_Float16><<<gp, blk, 0, stream>>>(xqh, Wqt, bq, Qh, M, HD_, D_);
  gemm_wmma<_Float16><<<gp, blk, 0, stream>>>(xkh, Wkt, bk, Kf, M, HD_, D_);
  gemm_wmma<_Float16><<<gp, blk, 0, stream>>>(xvh, Wvt, bv, Vf, M, HD_, D_);

  // block-diagonal attention
  block_attn<<<dim3(S_ / BLK_, H_, B_), blk, 0, stream>>>(Qh, Kf, Vf, mask, ctx);

  // output projection (f16 in, f32 out)
  dim3 go(D_ / 128, M / 128);
  gemm_wmma<float><<<go, blk, 0, stream>>>(ctx, Wot, bo, (float*)d_out, M, D_, HD_);
}
